// GraphNN_4063039062397
// MI455X (gfx1250) — compile-verified, matrix operand order checked by reading, not run on hardware
//
#include <hip/hip_runtime.h>
#include <hip/hip_bf16.h>

// ---------------------------------------------------------------------------
// DAG MLP on MI455X (gfx1250): 28 edge-GEMMs via v_wmma_f32_16x16x32_bf16.
// B strips staged in LDS (double-buffered, bank-conflict padded), A fragments
// software-pipelined; per-node global L2 norm via deterministic reductions.
// ---------------------------------------------------------------------------

typedef __attribute__((ext_vector_type(16))) __bf16 v16bf;
typedef __attribute__((ext_vector_type(8)))  float  v8f;

#define BATCH 32768
#define WIDTH 256
#define NELEM (BATCH * WIDTH)          // 8,388,608 elements per activation
#define HB    ((size_t)NELEM)          // bf16 activation stride (elements)
#define NEDGE 28
#define LDS_ROW 264                    // 256 + 8 ushort pad -> conflict-free b128 reads

union Frag { uint4 q[2]; v16bf v; };   // 32B fragment assembled from 2x16B LDS loads

// round-to-nearest-even f32 -> bf16 (bit pattern)
__device__ __forceinline__ unsigned short f2bf(float f) {
    unsigned u = __builtin_bit_cast(unsigned, f);
    u += 0x7fffu + ((u >> 16) & 1u);
    return (unsigned short)(u >> 16);
}

// ---- W[e][k][n] f32  ->  Wt[e][n][k] bf16 (transposed for WMMA B striping)
__global__ void wt_kernel(const float* __restrict__ W, unsigned short* __restrict__ Wt) {
    int idx = blockIdx.x * blockDim.x + threadIdx.x;     // over 28*256*256
    int e = idx >> 16;
    int r = idx & 0xffff;
    int n = r >> 8;
    int k = r & 255;
    float v = W[(size_t)e * (WIDTH * WIDTH) + (size_t)k * WIDTH + n];
    Wt[idx] = f2bf(v);
}

// ---- stage-1 sum of squares of a f32 tensor -> one partial per block
__global__ void sumsq_partial_kernel(const float* __restrict__ x,
                                     float* __restrict__ partials, int n) {
    float s = 0.f;
    for (int i = blockIdx.x * blockDim.x + threadIdx.x; i < n;
         i += gridDim.x * blockDim.x) {
        float v = x[i];
        s += v * v;
    }
    for (int off = 16; off > 0; off >>= 1) s += __shfl_xor(s, off, 32);
    __shared__ float sm[8];
    int lane = threadIdx.x & 31, wave = threadIdx.x >> 5;
    if (lane == 0) sm[wave] = s;
    __syncthreads();
    if (threadIdx.x == 0) {
        float t = 0.f;
        for (int w = 0; w < 8; ++w) t += sm[w];
        partials[blockIdx.x] = t;
    }
}

// ---- stage-2: deterministically sum block partials into *out
__global__ void reduce_finish_kernel(const float* __restrict__ partials, int n,
                                     float* __restrict__ out) {
    __shared__ float sm[256];
    float s = 0.f;
    for (int i = threadIdx.x; i < n; i += blockDim.x) s += partials[i];
    sm[threadIdx.x] = s;
    __syncthreads();
    for (int st = 128; st > 0; st >>= 1) {
        if (threadIdx.x < st) sm[threadIdx.x] += sm[threadIdx.x + st];
        __syncthreads();
    }
    if (threadIdx.x == 0) *out = sm[0];
}

// ---- h = src * rsqrt(ss); write bf16 copy (GEMM input) and optional f32 copy
__global__ void scale_kernel(const float* __restrict__ src,
                             unsigned short* __restrict__ dstbf,
                             float* __restrict__ dstf,
                             const float* __restrict__ ss, int n) {
    float inv = rsqrtf(ss[0]);
    for (int i = blockIdx.x * blockDim.x + threadIdx.x; i < n;
         i += gridDim.x * blockDim.x) {
        float v = src[i] * inv;
        dstbf[i] = f2bf(v);
        if (dstf) dstf[i] = v;
    }
}

// ---- node GEMM: acc = sum_{i<j} relu(h_i @ W_e + b_e), plus sumsq partials.
// 8 waves / block, each wave owns a 16(M) x 64(N) strip; all waves share the
// B strip (64 N x 256 K bf16 = 32KB) -> stage once per block in LDS, double-
// buffered across edges so B[e+1] global loads overlap edge-e WMMAs.
__global__ __launch_bounds__(256) void dag_gemm_kernel(
    const unsigned short* __restrict__ hbase,   // bf16 activations, stride HB
    const unsigned short* __restrict__ Wt,      // bf16 [E][N][K]
    const float* __restrict__ bias,             // f32 [E][N]
    float* __restrict__ acc,                    // f32 [BATCH][WIDTH] pre-norm out
    float* __restrict__ partials,               // sumsq partial per block
    int nEdges, int edgeBase) {
    __shared__ unsigned short Bs[2][64 * LDS_ROW];   // 2 x 33KB, padded rows
    __shared__ float smred[8];

    const int tid   = threadIdx.x;
    const int lane  = tid & 31;
    const int wave  = tid >> 5;
    const int m0    = (blockIdx.x * 8 + wave) * 16;
    const int n0    = blockIdx.y * 64;
    const int mrow  = m0 + (lane & 15);
    const int khalf = (lane >> 4) << 4;     // 0 or 16

    // ---- stage B[edge 0] into LDS buffer 0 (contiguous 32KB strip of Wt)
    uint4 stg[8];
    {
        const uint4* src = (const uint4*)(Wt + (size_t)edgeBase * (WIDTH * WIDTH)
                                             + (size_t)n0 * WIDTH);
#pragma unroll
        for (int t = 0; t < 8; ++t) stg[t] = src[tid + t * 256];
#pragma unroll
        for (int t = 0; t < 8; ++t) {
            int c = tid + t * 256;               // 2048 16B chunks
            int row = c >> 5, col = c & 31;      // 32 chunks per 256-ushort row
            *(uint4*)&Bs[0][row * LDS_ROW + col * 8] = stg[t];
        }
    }
    __syncthreads();

    v8f out0 = {}, out1 = {}, out2 = {}, out3 = {};

    for (int e = 0; e < nEdges; ++e) {
        // issue global loads for next edge's B strip (latency hidden by WMMAs)
        if (e + 1 < nEdges) {
            const uint4* src = (const uint4*)(Wt + (size_t)(edgeBase + e + 1) * (WIDTH * WIDTH)
                                                 + (size_t)n0 * WIDTH);
#pragma unroll
            for (int t = 0; t < 8; ++t) stg[t] = src[tid + t * 256];
        }

        const unsigned short* A  = hbase + (size_t)e * HB + (size_t)mrow * WIDTH + khalf;
        const unsigned short* bs = &Bs[e & 1][0];
        const int rowbase = (lane & 15) * LDS_ROW + khalf;

        v8f c0 = {}, c1 = {}, c2 = {}, c3 = {};
        v16bf a_cur = *(const v16bf*)(A);        // k0 = 0, software-pipelined A
#pragma unroll
        for (int k0 = 0; k0 < WIDTH; k0 += 32) {
            v16bf a_next = a_cur;
            if (k0 + 32 < WIDTH) a_next = *(const v16bf*)(A + k0 + 32);

            Frag b0, b1, b2, b3;
            const unsigned short* bp = bs + rowbase + k0;
            b0.q[0] = *(const uint4*)(bp);                    b0.q[1] = *(const uint4*)(bp + 8);
            b1.q[0] = *(const uint4*)(bp + 16 * LDS_ROW);     b1.q[1] = *(const uint4*)(bp + 16 * LDS_ROW + 8);
            b2.q[0] = *(const uint4*)(bp + 32 * LDS_ROW);     b2.q[1] = *(const uint4*)(bp + 32 * LDS_ROW + 8);
            b3.q[0] = *(const uint4*)(bp + 48 * LDS_ROW);     b3.q[1] = *(const uint4*)(bp + 48 * LDS_ROW + 8);

            c0 = __builtin_amdgcn_wmma_f32_16x16x32_bf16(false, a_cur, false, b0.v, (short)0, c0, false, false);
            c1 = __builtin_amdgcn_wmma_f32_16x16x32_bf16(false, a_cur, false, b1.v, (short)0, c1, false, false);
            c2 = __builtin_amdgcn_wmma_f32_16x16x32_bf16(false, a_cur, false, b2.v, (short)0, c2, false, false);
            c3 = __builtin_amdgcn_wmma_f32_16x16x32_bf16(false, a_cur, false, b3.v, (short)0, c3, false, false);
            a_cur = a_next;
        }

        const float* bv = bias + (size_t)(edgeBase + e) * WIDTH + n0 + (lane & 15);
        float bb0 = bv[0], bb1 = bv[16], bb2 = bv[32], bb3 = bv[48];
#pragma unroll
        for (int r = 0; r < 8; ++r) {
            out0[r] += fmaxf(c0[r] + bb0, 0.f);
            out1[r] += fmaxf(c1[r] + bb1, 0.f);
            out2[r] += fmaxf(c2[r] + bb2, 0.f);
            out3[r] += fmaxf(c3[r] + bb3, 0.f);
        }

        // commit staged B[e+1] into the other LDS buffer
        if (e + 1 < nEdges) {
            __syncthreads();   // all waves finished reading buffer (e+1)&1 (edge e-1)
#pragma unroll
            for (int t = 0; t < 8; ++t) {
                int c = tid + t * 256;
                int row = c >> 5, col = c & 31;
                *(uint4*)&Bs[(e + 1) & 1][row * LDS_ROW + col * 8] = stg[t];
            }
            __syncthreads();
        }
    }

    // C/D layout: VGPR r, lanes 0-15 -> row m0+r, lanes 16-31 -> row m0+r+8
    float ss = 0.f;
#pragma unroll
    for (int r = 0; r < 8; ++r) {
        int rr = m0 + ((lane < 16) ? r : r + 8);
        size_t base = (size_t)rr * WIDTH + n0 + (lane & 15);
        acc[base     ] = out0[r]; ss += out0[r] * out0[r];
        acc[base + 16] = out1[r]; ss += out1[r] * out1[r];
        acc[base + 32] = out2[r]; ss += out2[r] * out2[r];
        acc[base + 48] = out3[r]; ss += out3[r] * out3[r];
    }
    for (int off = 16; off > 0; off >>= 1) ss += __shfl_xor(ss, off, 32);
    if (lane == 0) smred[wave] = ss;
    __syncthreads();
    if (tid == 0) {
        float t = 0.f;
        for (int w = 0; w < 8; ++w) t += smred[w];
        partials[blockIdx.y * gridDim.x + blockIdx.x] = t;
    }
}

// ---------------------------------------------------------------------------
extern "C" void kernel_launch(void* const* d_in, const int* in_sizes, int n_in,
                              void* d_out, int out_size, void* d_ws, size_t ws_size,
                              hipStream_t stream) {
    const float* x = (const float*)d_in[0];   // [32768,256]
    const float* W = (const float*)d_in[1];   // [28,256,256]
    const float* b = (const float*)d_in[2];   // [28,256]
    float* out = (float*)d_out;               // [32768,256]

    char* ws = (char*)d_ws;
    const size_t WT_BYTES  = (size_t)NEDGE * WIDTH * WIDTH * sizeof(unsigned short); // 3.67MB
    unsigned short* Wt     = (unsigned short*)(ws);
    float*          sums   = (float*)(ws + WT_BYTES);            // 16 floats
    float*          parts  = sums + 16;                          // block partials
    float*          acc    = (float*)(ws + (4ull << 20));        // 32MB @ 4MB
    unsigned short* hbf    = (unsigned short*)(ws + (40ull << 20)); // 8x16MB @ 40MB

    // 1) convert + transpose weights to bf16 [E][N][K]
    wt_kernel<<<(NEDGE * WIDTH * WIDTH) / 256, 256, 0, stream>>>(W, Wt);

    // 2) h0 = l2norm(x), stored as bf16
    sumsq_partial_kernel<<<2048, 256, 0, stream>>>(x, parts, NELEM);
    reduce_finish_kernel<<<1, 256, 0, stream>>>(parts, 2048, &sums[0]);
    scale_kernel<<<2048, 256, 0, stream>>>(x, hbf, nullptr, &sums[0], NELEM);

    // 3) nodes 1..7: fused edge-GEMMs + relu-sum, then global norm
    dim3 grid(BATCH / (16 * 8), WIDTH / 64);   // 256 x 4 blocks
    for (int j = 1; j < 8; ++j) {
        int eb = j * (j - 1) / 2;              // edge (i,j) -> eb + i
        dag_gemm_kernel<<<grid, 256, 0, stream>>>(hbf, Wt, b, acc, parts, j, eb);
        reduce_finish_kernel<<<1, 256, 0, stream>>>(parts, grid.x * grid.y, &sums[j]);
        // last node: l2norm(l2norm(acc)) == l2norm(acc) (unit norm is idempotent)
        scale_kernel<<<2048, 256, 0, stream>>>(acc, hbf + (size_t)j * HB,
                                               (j == 7) ? out : nullptr,
                                               &sums[j], NELEM);
    }
}
